// DeepSNN_64819646431520
// MI455X (gfx1250) — matrix operand order repaired and checked by reading, not compiled, not run on hardware
//
#include <hip/hip_runtime.h>

// ---------------------------------------------------------------------------
// DeepSNN on MI455X (gfx1250, wave32):
//   GEMM0 (spikes@W0) -> LIF+LN1 -> GEMM1 -> LIF+LN2 -> GEMM2 -> LIF+pool
//   -> finalize (global mean + 64x2 classifier)
//
// GEMM: v_wmma_f32_16x16x32_bf16 with DOUBLE-BUFFERED Tensor-Data-Mover
// streaming of A tiles (tensor_load_to_lds + s_wait_tensorcnt): wave 0 issues
// the DMA for chunk c+1, all waves compute chunk c, one barrier per chunk.
// W panel staged in LDS once per block (transposed, padded rows).
// ---------------------------------------------------------------------------

typedef __attribute__((ext_vector_type(16))) __bf16 bf16x16;
typedef __attribute__((ext_vector_type(8)))  float  f32x8;
typedef __attribute__((ext_vector_type(4)))  unsigned int u32x4;
typedef __attribute__((ext_vector_type(8)))  unsigned int u32x8;

#define ALPHA_MEM 0.9512294245007140f   // exp(-1/20)
#define ALPHA_SYN 0.8187307530779818f   // exp(-1/5)
#define ONE_MINUS_AMEM 0.0487705754992860f
#define LN_EPS 1e-6f

// ---------------------------------------------------------------------------
// f32 -> bf16 conversion (weights)
// ---------------------------------------------------------------------------
__global__ __launch_bounds__(256) void cvt_bf16_kernel(
    const float* __restrict__ in, __bf16* __restrict__ out, int n) {
  int i = blockIdx.x * 256 + threadIdx.x;
  if (i < n) out[i] = (__bf16)in[i];
}

// ---------------------------------------------------------------------------
// WMMA GEMM, TDM-pipelined A stream.
//   C[M,N] = A[M,K] @ W[K,N] + bias      (A f32|bf16, W bf16, C f32)
// Block: 256 threads = 8 wave32s; tile 64(M) x 32(N); waves 4(m) x 2(n).
//
// LDS pool:
//   [0, 2*A_BYTES):  double-buffered A chunks, 64 rows x (KT elems + 16B pad)
//                    (TDM pad feature: interval = row dwords, amount = 4 dw)
//   [2*A_BYTES, ..): W panel transposed [32 n][K+8] bf16, staged once
// ---------------------------------------------------------------------------
template <bool A_F32, int K>
__global__ __launch_bounds__(256) void gemm_wmma_tdm_kernel(
    const void* __restrict__ Av, const __bf16* __restrict__ W,
    const float* __restrict__ bias, float* __restrict__ C,
    int M, int N) {
  constexpr int BM = 64, BN = 32;
  constexpr int KT = A_F32 ? 32 : 128;      // K-extent of one A chunk
  constexpr int EA = A_F32 ? 4 : 2;         // A element size
  constexpr int ROW_DW = KT * EA / 4;       // 32 or 64 dwords per row
  constexpr unsigned PAD_IVL = (ROW_DW == 32) ? 4u : 5u;  // 2^(c+1) dwords
  constexpr int LDA_BYTES = KT * EA + 16;   // 144 or 272 (16B pad per row)
  constexpr int A_BYTES = BM * LDA_BYTES;
  constexpr int LDB = K + 8;                // bf16 elems (rows 16B-aligned)
  constexpr int B_OFF = 2 * A_BYTES;
  constexpr int NCHUNK = K / KT;
  __shared__ __align__(16) char pool[B_OFF + BN * LDB * 2];

  const int tid  = threadIdx.x;
  const int lane = tid & 31;
  const int wave = tid >> 5;
  const int wm   = wave & 3;                // 4 m-tiles
  const int wn   = wave >> 2;               // 2 n-tiles
  const int m0   = blockIdx.x * BM;
  const int n0   = blockIdx.y * BN;

  // ---- one-time W panel staging (transposed: Bls[n][k]) ----
  {
    __bf16* Bls = (__bf16*)(pool + B_OFF);
    for (int i = tid; i < K * BN; i += 256) {
      const int k = i >> 5, n = i & 31;     // 32 consecutive n per k: coalesced
      Bls[n * LDB + k] = W[(size_t)k * N + n0 + n];
    }
  }

  f32x8 acc = {0.f, 0.f, 0.f, 0.f, 0.f, 0.f, 0.f, 0.f};

  const int a_row = wm * 16 + (lane & 15);
  const int a_kb  = (lane >> 4) * 8;
  const int b_col = wn * 16 + (lane & 15);
  const int b_kb  = (lane >> 4) * 8;
  const __bf16* Bls = (const __bf16*)(pool + B_OFF);

  // LDS aperture: flat addr low 32 bits == wave-relative LDS byte address
  const unsigned lds_base = (unsigned)(size_t)&pool[0];
  const unsigned long long abase =
      (unsigned long long)(size_t)Av + (size_t)m0 * K * EA;

  // D# issue helper: 2D tile [BM rows x KT cols] of A into LDS buffer `buf`
  auto tdm_issue = [&](int chunk, int buf) {
    const unsigned long long ga =
        abase + (unsigned long long)chunk * KT * EA;
    u32x4 g0;
    g0[0] = 1u;                                        // count=1, user mode
    g0[1] = lds_base + (unsigned)(buf * A_BYTES);      // D#.lds_addr
    g0[2] = (unsigned)ga;                              // D#.global_addr lo
    g0[3] = (unsigned)((ga >> 32) & 0x01FFFFFFu)       // D#.global_addr hi
          | (2u << 30);                                // D#.type = 2 (image)
    u32x8 g1;
    g1[0] = ((A_F32 ? 2u : 1u) << 16)                  // data_size (4B/2B)
          | (1u << 20)                                 // pad_enable
          | (PAD_IVL << 22)                            // pad_interval = row
          | (3u << 25);                                // pad_amount: 4 dwords
    g1[1] = ((unsigned)K & 0xFFFFu) << 16;             // tensor_dim0 [15:0]
    g1[2] = (((unsigned)K >> 16) & 0xFFFFu)            // tensor_dim0 [31:16]
          | (((unsigned)M & 0xFFFFu) << 16);           // tensor_dim1 [15:0]
    g1[3] = (((unsigned)M >> 16) & 0xFFFFu)            // tensor_dim1 [31:16]
          | ((unsigned)KT << 16);                      // tile_dim0 = KT
    g1[4] = (unsigned)BM;                              // tile_dim1 (dim2=0)
    g1[5] = (unsigned)K;                               // dim0_stride [31:0]
    g1[6] = 0u;                                        // stride hi bits
    g1[7] = 0u;
    asm volatile("tensor_load_to_lds %0, %1" :: "s"(g0), "s"(g1) : "memory");
  };

  // ---- pipeline prologue: fetch chunk 0 into buffer 0 ----
  if (wave == 0) {
    tdm_issue(0, 0);
    __builtin_amdgcn_s_wait_tensorcnt(0);
  }
  __syncthreads();   // publish A chunk 0 + W panel

  int buf = 0;
  for (int c = 0; c < NCHUNK; ++c) {
    // ---- kick off DMA for the NEXT chunk while we compute this one ----
    if (wave == 0 && (c + 1) < NCHUNK) tdm_issue(c + 1, buf ^ 1);

    const char* abuf = pool + buf * A_BYTES;
#pragma unroll
    for (int kk = 0; kk < KT; kk += 32) {
      union { bf16x16 v; float4 f4[2]; __bf16 e[16]; } fa, fb;
      if (A_F32) {
        const float* arow = (const float*)(abuf + a_row * LDA_BYTES + kk * 4);
        float t[16];
        *(float4*)&t[0]  = *(const float4*)(arow + a_kb);
        *(float4*)&t[4]  = *(const float4*)(arow + a_kb + 4);
        *(float4*)&t[8]  = *(const float4*)(arow + a_kb + 16);
        *(float4*)&t[12] = *(const float4*)(arow + a_kb + 20);
#pragma unroll
        for (int i = 0; i < 16; ++i) fa.e[i] = (__bf16)t[i];  // v_cvt_pk_bf16
      } else {
        const __bf16* arow = (const __bf16*)(abuf + a_row * LDA_BYTES + kk * 2);
        fa.f4[0] = *(const float4*)(arow + a_kb);
        fa.f4[1] = *(const float4*)(arow + 16 + a_kb);
      }
      const __bf16* brow = &Bls[b_col * LDB + c * KT + kk + b_kb];
      fb.f4[0] = *(const float4*)(brow);
      fb.f4[1] = *(const float4*)(brow + 16);

      acc = __builtin_amdgcn_wmma_f32_16x16x32_bf16(
          /*neg_a=*/false, fa.v, /*neg_b=*/false, fb.v,
          /*c_mod=*/(short)0, acc, /*reuse_a=*/false, /*reuse_b=*/false);
    }

    // ---- retire this chunk, publish the next (TDM done -> barrier) ----
    if (wave == 0) __builtin_amdgcn_s_wait_tensorcnt(0);
    __syncthreads();
    buf ^= 1;
  }

  // ---- epilogue: C layout (lanes 0-15: M=r, lanes 16-31: M=8+r; N=lane&15)
  const int crow0 = m0 + wm * 16 + (lane >> 4) * 8;
  const int ccol  = n0 + wn * 16 + (lane & 15);
  const float bv = bias[ccol];
#pragma unroll
  for (int r = 0; r < 8; ++r)
    C[(size_t)(crow0 + r) * N + ccol] = acc[r] + bv;
}

// ---------------------------------------------------------------------------
// Fused LIF scan + LayerNorm: one block per batch sample, one thread per h.
// ---------------------------------------------------------------------------
template <int H>
__global__ __launch_bounds__(H) void lif_ln_kernel(
    const float* __restrict__ cur,        // [B*T, H]
    const float* __restrict__ ln_scale, const float* __restrict__ ln_bias,
    __bf16* __restrict__ out,             // [B*T, H]
    int T) {
  constexpr int NW = H / 32;
  const int b = blockIdx.x;
  const int h = threadIdx.x;
  const int lane = h & 31;
  const int wave = h >> 5;
  __shared__ float2 red[NW];
  __shared__ float2 stats;

  float vm = 0.f, cs = 0.f;
  const float sc = ln_scale[h], bi = ln_bias[h];
  const size_t base = (size_t)b * T * H + h;
  const float invH = 1.f / (float)H;

  for (int t = 0; t < T; ++t) {
    const float ic = cur[base + (size_t)t * H];
    __builtin_prefetch(&cur[base + (size_t)(t + 4) * H], 0, 1);
    cs = ALPHA_SYN * cs + ic;
    vm = ALPHA_MEM * vm + ONE_MINUS_AMEM * cs;
    const float spike = (vm >= 1.f) ? 1.f : 0.f;
    const float bx = 4.f * (vm - 1.f);
    // sg = BETA / (2*(1+cosh(bx))) = 4 / (2 + e^bx + e^-bx)
    const float sg = 4.f / (2.f + __expf(bx) + __expf(-bx));
    const float swg = 2.f * spike - sg;   // fwd value of surrogate expression
    vm = (spike > 0.f) ? 0.f : vm;

    float sx = swg, sxx = swg * swg;
#pragma unroll
    for (int off = 16; off > 0; off >>= 1) {
      sx  += __shfl_xor(sx, off);
      sxx += __shfl_xor(sxx, off);
    }
    if (lane == 0) red[wave] = make_float2(sx, sxx);
    __syncthreads();
    if (h == 0) {
      float tx = 0.f, txx = 0.f;
#pragma unroll
      for (int w = 0; w < NW; ++w) { tx += red[w].x; txx += red[w].y; }
      const float mu  = tx * invH;
      const float var = txx * invH - mu * mu;
      stats = make_float2(mu, rsqrtf(var + LN_EPS));
    }
    __syncthreads();
    const float mu = stats.x, rs = stats.y;
    out[base + (size_t)t * H] = (__bf16)((swg - mu) * rs * sc + bi);
  }
}

// ---------------------------------------------------------------------------
// Final LIF layer (H=64) + temporal mean pooling + per-sample partial sums.
// ---------------------------------------------------------------------------
__global__ __launch_bounds__(64) void lif_pool_kernel(
    const float* __restrict__ cur,       // [B*T, 64]
    float* __restrict__ pooled,          // [B, 64]  (into d_out)
    float* __restrict__ partial,         // [B]      (ws scratch)
    int T) {
  const int b = blockIdx.x;
  const int h = threadIdx.x;
  float vm = 0.f, cs = 0.f, accum = 0.f;
  const size_t base = (size_t)b * T * 64 + h;
  for (int t = 0; t < T; ++t) {
    const float ic = cur[base + (size_t)t * 64];
    __builtin_prefetch(&cur[base + (size_t)(t + 4) * 64], 0, 1);
    cs = ALPHA_SYN * cs + ic;
    vm = ALPHA_MEM * vm + ONE_MINUS_AMEM * cs;
    const float spike = (vm >= 1.f) ? 1.f : 0.f;
    const float bx = 4.f * (vm - 1.f);
    const float sg = 4.f / (2.f + __expf(bx) + __expf(-bx));
    accum += 2.f * spike - sg;
    vm = (spike > 0.f) ? 0.f : vm;
  }
  pooled[b * 64 + h] = accum / (float)T;

  __shared__ float red[2];
  float p = accum;
#pragma unroll
  for (int off = 16; off > 0; off >>= 1) p += __shfl_xor(p, off);
  if ((h & 31) == 0) red[h >> 5] = p;
  __syncthreads();
  if (h == 0) partial[b] = red[0] + red[1];
}

// ---------------------------------------------------------------------------
// Global mean over x2 + classifier logits = pooled @ Wc + bc  (single block)
// ---------------------------------------------------------------------------
__global__ __launch_bounds__(256) void finalize_kernel(
    const float* __restrict__ partial,   // [512]
    const float* __restrict__ pooled,    // [512, 64]
    const float* __restrict__ Wc,        // [64, 2]
    const float* __restrict__ bc,        // [2]
    float* __restrict__ logits,          // [512, 2]
    float* __restrict__ mean_out) {
  const int tid = threadIdx.x;
  __shared__ float red[8];
  float s = partial[tid] + partial[tid + 256];
#pragma unroll
  for (int off = 16; off > 0; off >>= 1) s += __shfl_xor(s, off);
  if ((tid & 31) == 0) red[tid >> 5] = s;
  __syncthreads();
  if (tid == 0) {
    float tot = 0.f;
#pragma unroll
    for (int w = 0; w < 8; ++w) tot += red[w];
    mean_out[0] = tot / (512.f * 128.f * 64.f);
  }
  for (int b = tid; b < 512; b += 256) {
    float l0 = bc[0], l1 = bc[1];
    for (int hh = 0; hh < 64; ++hh) {
      const float p = pooled[b * 64 + hh];
      l0 += p * Wc[hh * 2 + 0];
      l1 += p * Wc[hh * 2 + 1];
    }
    logits[b * 2 + 0] = l0;
    logits[b * 2 + 1] = l1;
  }
}

// ---------------------------------------------------------------------------
extern "C" void kernel_launch(void* const* d_in, const int* in_sizes, int n_in,
                              void* d_out, int out_size, void* d_ws, size_t ws_size,
                              hipStream_t stream) {
  (void)in_sizes; (void)n_in; (void)out_size; (void)ws_size;
  const float* spikes = (const float*)d_in[0];   // [512,128,512]
  const float* W0     = (const float*)d_in[1];   // [512,256]
  const float* b0     = (const float*)d_in[2];
  const float* ln1s   = (const float*)d_in[3];
  const float* ln1b   = (const float*)d_in[4];
  const float* W1     = (const float*)d_in[5];   // [256,128]
  const float* b1     = (const float*)d_in[6];
  const float* ln2s   = (const float*)d_in[7];
  const float* ln2b   = (const float*)d_in[8];
  const float* W2     = (const float*)d_in[9];   // [128,64]
  const float* b2     = (const float*)d_in[10];
  const float* Wc     = (const float*)d_in[11];  // [64,2]
  const float* bc     = (const float*)d_in[12];

  float* out = (float*)d_out;     // logits[1024] | pooled[32768] | mean[1]
  float* out_logits = out;
  float* out_pooled = out + 1024;
  float* out_mean   = out + 33792;

  const int B = 512, T = 128, D = 512, H0 = 256, H1 = 128, H2 = 64;
  const int M = B * T;            // 65536

  // workspace bump allocation (256B aligned)
  char* wsp = (char*)d_ws;
  auto walloc = [&](size_t bytes) {
    char* p = wsp;
    wsp += (bytes + 255) & ~(size_t)255;
    return p;
  };
  __bf16* w0b = (__bf16*)walloc((size_t)D * H0 * 2);
  __bf16* w1b = (__bf16*)walloc((size_t)H0 * H1 * 2);
  __bf16* w2b = (__bf16*)walloc((size_t)H1 * H2 * 2);
  float*  cur0 = (float*)walloc((size_t)M * H0 * 4);
  __bf16* a1   = (__bf16*)walloc((size_t)M * H0 * 2);
  float*  cur1 = (float*)walloc((size_t)M * H1 * 4);
  __bf16* a2   = (__bf16*)walloc((size_t)M * H1 * 2);
  float*  cur2 = (float*)walloc((size_t)M * H2 * 4);
  float*  partial = (float*)walloc((size_t)B * 4);

  // weight conversion to bf16
  cvt_bf16_kernel<<<(D * H0 + 255) / 256, 256, 0, stream>>>(W0, w0b, D * H0);
  cvt_bf16_kernel<<<(H0 * H1 + 255) / 256, 256, 0, stream>>>(W1, w1b, H0 * H1);
  cvt_bf16_kernel<<<(H1 * H2 + 255) / 256, 256, 0, stream>>>(W2, w2b, H1 * H2);

  // layer 0: GEMM (A f32, TDM-pipelined + converted) + LIF + LN1
  gemm_wmma_tdm_kernel<true, 512><<<dim3(M / 64, H0 / 32), 256, 0, stream>>>(
      spikes, w0b, b0, cur0, M, H0);
  lif_ln_kernel<256><<<B, 256, 0, stream>>>(cur0, ln1s, ln1b, a1, T);

  // layer 1: GEMM + LIF + LN2
  gemm_wmma_tdm_kernel<false, 256><<<dim3(M / 64, H1 / 32), 256, 0, stream>>>(
      a1, w1b, b1, cur1, M, H1);
  lif_ln_kernel<128><<<B, 128, 0, stream>>>(cur1, ln2s, ln2b, a2, T);

  // layer 2: GEMM + LIF + pooling
  gemm_wmma_tdm_kernel<false, 128><<<dim3(M / 64, H2 / 32), 256, 0, stream>>>(
      a2, w2b, b2, cur2, M, H2);
  lif_pool_kernel<<<B, 64, 0, stream>>>(cur2, out_pooled, partial, T);

  // global mean + classifier
  finalize_kernel<<<1, 256, 0, stream>>>(partial, out_pooled, Wc, bc,
                                         out_logits, out_mean);
}
